// MultiHeadAttention_46050639348198
// MI455X (gfx1250) — matrix operand order repaired and checked by reading, not compiled, not run on hardware
//
#include <hip/hip_runtime.h>

// ---------------------------------------------------------------- types
typedef __attribute__((ext_vector_type(16))) __bf16 v16bf;
typedef __attribute__((ext_vector_type(8)))  __bf16 v8bf;
typedef __attribute__((ext_vector_type(8)))  float  v8f;

static constexpr int Bb = 2, Ss = 2048, Dd = 1024, Hh = 16, DH = 64;
static constexpr float MASKEDF = -2147483648.0f;   // -2^31 per reference

// float -> bf16, round-to-nearest-even
__device__ __forceinline__ __bf16 f2bf(float f) {
  unsigned u = __builtin_bit_cast(unsigned, f);
  u += 0x7FFFu + ((u >> 16) & 1u);
  unsigned short h = (unsigned short)(u >> 16);
  return __builtin_bit_cast(__bf16, h);
}

__device__ __forceinline__ v16bf cat8(v8bf lo, v8bf hi) {
  v16bf a;
#pragma unroll
  for (int i = 0; i < 8; ++i) { a[i] = lo[i]; a[i + 8] = hi[i]; }
  return a;
}

// ---------------------------------------------------------------- 1) f32 -> bf16
__global__ void cvt_bf16_kernel(const float* __restrict__ src,
                                __bf16* __restrict__ dst, int n) {
  int i = (blockIdx.x * blockDim.x + threadIdx.x) * 4;
  if (i + 3 < n) {
    float4 v = *(const float4*)(src + i);
    dst[i + 0] = f2bf(v.x); dst[i + 1] = f2bf(v.y);
    dst[i + 2] = f2bf(v.z); dst[i + 3] = f2bf(v.w);
  }
}

// ---------------------------------------------------------------- 2) Wq -> WqT (bf16), WqT[k*D+n] = Wq[n*D+k]
__global__ void wqt_kernel(const float* __restrict__ Wq, __bf16* __restrict__ WqT) {
  int i = blockIdx.x * blockDim.x + threadIdx.x;     // over D*D
  int k = i >> 10, n = i & (Dd - 1);
  WqT[i] = f2bf(Wq[(size_t)n * Dd + k]);
}

// ---------------------------------------------------------------- 3) P = X * WqT + bq  (M x 1024, K = 1024), bf16 WMMA
__global__ void proj_wmma_kernel(const __bf16* __restrict__ X,
                                 const __bf16* __restrict__ WqT,
                                 const float* __restrict__ bias,
                                 __bf16* __restrict__ P, int M) {
  int lane = threadIdx.x & 31;
  int wave = blockIdx.x * (blockDim.x >> 5) + (threadIdx.x >> 5);
  const int ntn = Dd / 16;                            // 64 tiles along N
  int tm = (wave / ntn) * 16;
  int tn = (wave % ntn) * 16;
  if (tm >= M) return;
  int r = lane & 15, hi = lane >> 4;

  const __bf16* arow = X + (size_t)(tm + r) * Dd;     // lane's A row
  v8f acc = {};
  for (int k0 = 0; k0 < Dd; k0 += 32) {
    __builtin_prefetch(arow + k0 + 64);               // global_prefetch_b8
    v8bf a0 = *(const v8bf*)(arow + k0 + hi * 8);
    v8bf a1 = *(const v8bf*)(arow + k0 + 16 + hi * 8);
    v16bf a = cat8(a0, a1);
    v16bf b;
#pragma unroll
    for (int i = 0; i < 8; ++i) {
      b[i]     = WqT[(size_t)(k0 + hi * 8 + i) * Dd + tn + r];
      b[i + 8] = WqT[(size_t)(k0 + 16 + hi * 8 + i) * Dd + tn + r];
    }
    acc = __builtin_amdgcn_wmma_f32_16x16x32_bf16(false, a, false, b,
                                                  (short)0, acc, false, false);
  }
  float bv = bias[tn + r];
#pragma unroll
  for (int rr = 0; rr < 8; ++rr) {
    int m = tm + hi * 8 + rr;
    P[(size_t)m * Dd + tn + r] = f2bf(acc[rr] + bv);
  }
}

// ---------------------------------------------------------------- 4) kp[B,S,H,64] -> kpT[B,H,64,S]
__global__ void kpt_kernel(const __bf16* __restrict__ kp, __bf16* __restrict__ kpT) {
  size_t i = (size_t)blockIdx.x * blockDim.x + threadIdx.x;  // 4,194,304 total
  int ki = (int)(i & (Ss - 1));
  int d  = (int)((i >> 11) & 63);
  int h  = (int)((i >> 17) & 15);
  int b  = (int)(i >> 21);
  kpT[i] = kp[((size_t)(b * Ss + ki)) * Dd + h * DH + d];
}

// ---------------------------------------------------------------- 5) e = (qp . kp^T)/8, mask -> -2^31
__global__ void scores_wmma_kernel(const __bf16* __restrict__ qp,
                                   const __bf16* __restrict__ kpT,
                                   const int* __restrict__ kmask,
                                   float* __restrict__ e) {
  const int NT = Ss / 16;                              // 128
  int lane = threadIdx.x & 31;
  int wave = blockIdx.x * (blockDim.x >> 5) + (threadIdx.x >> 5);
  int tk = wave % NT;
  int tq = (wave / NT) % NT;
  int h  = (wave / (NT * NT)) % Hh;
  int b  =  wave / (NT * NT * Hh);
  int r = lane & 15, hi = lane >> 4;

  const __bf16* arow = qp + ((size_t)(b * Ss + tq * 16 + r)) * Dd + h * DH;
  v8f acc = {};
#pragma unroll
  for (int k0 = 0; k0 < DH; k0 += 32) {
    v8bf a0 = *(const v8bf*)(arow + k0 + hi * 8);
    v8bf a1 = *(const v8bf*)(arow + k0 + 16 + hi * 8);
    v16bf a = cat8(a0, a1);
    v16bf bm;
#pragma unroll
    for (int i = 0; i < 8; ++i) {
      bm[i]     = kpT[((size_t)((b * Hh + h) * DH + k0 + hi * 8 + i)) * Ss + tk * 16 + r];
      bm[i + 8] = kpT[((size_t)((b * Hh + h) * DH + k0 + 16 + hi * 8 + i)) * Ss + tk * 16 + r];
    }
    acc = __builtin_amdgcn_wmma_f32_16x16x32_bf16(false, a, false, bm,
                                                  (short)0, acc, false, false);
  }
  int col = tk * 16 + r;
  int km = kmask[b * Ss + col];
#pragma unroll
  for (int rr = 0; rr < 8; ++rr) {
    int qrow = tq * 16 + hi * 8 + rr;
    size_t idx = (((size_t)(b * Hh + h)) * Ss + qrow) * Ss + col;
    float val = acc[rr] * 0.125f;                      // 1/sqrt(64)
    e[idx] = (km == 0) ? MASKEDF : val;
  }
}

// ---------------------------------------------------------------- 6) per-row softmax stats over e
__global__ void softmax_stats_kernel(const float* __restrict__ e,
                                     float* __restrict__ rmax,
                                     float* __restrict__ rsum) {
  __shared__ float sm[256];
  int row = blockIdx.x;
  const float* er = e + (size_t)row * Ss;
  int t = threadIdx.x;
  float mx = -3.4e38f;
  for (int i = t; i < Ss; i += 256) mx = fmaxf(mx, er[i]);
  sm[t] = mx; __syncthreads();
  for (int s = 128; s > 0; s >>= 1) { if (t < s) sm[t] = fmaxf(sm[t], sm[t + s]); __syncthreads(); }
  float m = sm[0]; __syncthreads();
  float ssum = 0.f;
  for (int i = t; i < Ss; i += 256) ssum += __expf(er[i] - m);
  sm[t] = ssum; __syncthreads();
  for (int s = 128; s > 0; s >>= 1) { if (t < s) sm[t] += sm[t + s]; __syncthreads(); }
  if (t == 0) { rmax[row] = m; rsum[row] = sm[0]; }
}

// ---------------------------------------------------------------- 7) o = softmax(e) @ vp, q_mask gate
__global__ void av_wmma_kernel(const float* __restrict__ e,
                               const __bf16* __restrict__ vp,
                               const float* __restrict__ rmax,
                               const float* __restrict__ rsum,
                               const int* __restrict__ qmask,
                               float* __restrict__ o) {
  const int NT = Ss / 16;
  int lane = threadIdx.x & 31;
  int wave = blockIdx.x * (blockDim.x >> 5) + (threadIdx.x >> 5);
  int tq = wave % NT;
  int h  = (wave / NT) % Hh;
  int b  =  wave / (NT * Hh);
  int r = lane & 15, hi = lane >> 4;

  size_t row = ((size_t)(b * Hh + h)) * Ss + tq * 16 + r;   // lane's q row
  const float* erow = e + row * Ss;
  float rm = rmax[row];
  float rs = 1.0f / rsum[row];

  v8f acc0 = {}, acc1 = {}, acc2 = {}, acc3 = {};
  for (int k0 = 0; k0 < Ss; k0 += 32) {
    v8f e0 = *(const v8f*)(erow + k0 + hi * 8);
    v8f e1 = *(const v8f*)(erow + k0 + 16 + hi * 8);
    v16bf a;
#pragma unroll
    for (int i = 0; i < 8; ++i) {
      a[i]     = f2bf(__expf(e0[i] - rm) * rs);
      a[i + 8] = f2bf(__expf(e1[i] - rm) * rs);
    }
    const __bf16* vcol = vp + ((size_t)(b * Ss + k0)) * Dd + h * DH + r;
    v16bf b0, b1, b2, b3;
#pragma unroll
    for (int i = 0; i < 8; ++i) {
      size_t o0 = (size_t)(hi * 8 + i) * Dd, o1 = (size_t)(16 + hi * 8 + i) * Dd;
      b0[i] = vcol[o0];      b0[i + 8] = vcol[o1];
      b1[i] = vcol[o0 + 16]; b1[i + 8] = vcol[o1 + 16];
      b2[i] = vcol[o0 + 32]; b2[i + 8] = vcol[o1 + 32];
      b3[i] = vcol[o0 + 48]; b3[i + 8] = vcol[o1 + 48];
    }
    acc0 = __builtin_amdgcn_wmma_f32_16x16x32_bf16(false, a, false, b0, (short)0, acc0, false, false);
    acc1 = __builtin_amdgcn_wmma_f32_16x16x32_bf16(false, a, false, b1, (short)0, acc1, false, false);
    acc2 = __builtin_amdgcn_wmma_f32_16x16x32_bf16(false, a, false, b2, (short)0, acc2, false, false);
    acc3 = __builtin_amdgcn_wmma_f32_16x16x32_bf16(false, a, false, b3, (short)0, acc3, false, false);
  }
#pragma unroll
  for (int rr = 0; rr < 8; ++rr) {
    int m = tq * 16 + hi * 8 + rr;
    float gate = (qmask[b * Ss + m] != 0) ? 1.0f : 0.0f;
    size_t ob = ((size_t)(b * Ss + m)) * Dd + h * DH + r;
    o[ob]      = acc0[rr] * gate;
    o[ob + 16] = acc1[rr] * gate;
    o[ob + 32] = acc2[rr] * gate;
    o[ob + 48] = acc3[rr] * gate;
  }
}

// ---------------------------------------------------------------- launch
extern "C" void kernel_launch(void* const* d_in, const int* in_sizes, int n_in,
                              void* d_out, int out_size, void* d_ws, size_t ws_size,
                              hipStream_t stream) {
  const float* q  = (const float*)d_in[0];
  const float* k  = (const float*)d_in[1];
  const float* v  = (const float*)d_in[2];
  const float* Wq = (const float*)d_in[3];
  const float* bq = (const float*)d_in[4];
  const int* kmask = (const int*)d_in[5];
  const int* qmask = (const int*)d_in[6];

  float* o = (float*)d_out;                         // [B,S,D]
  float* e = o + (size_t)Bb * Ss * Dd;              // [B,H,S,S]

  // workspace carve-up (bytes)
  char* ws = (char*)d_ws;
  const size_t nBSD  = (size_t)Bb * Ss * Dd;        // 4,194,304 elems
  const size_t bfBSD = nBSD * sizeof(__bf16);       // 8 MiB
  __bf16* qb  = (__bf16*)(ws);                 ws += bfBSD;
  __bf16* kb  = (__bf16*)(ws);                 ws += bfBSD;
  __bf16* vb  = (__bf16*)(ws);                 ws += bfBSD;
  __bf16* WqT = (__bf16*)(ws);                 ws += (size_t)Dd * Dd * sizeof(__bf16);
  __bf16* qp  = (__bf16*)(ws);                 ws += bfBSD;
  __bf16* kp  = (__bf16*)(ws);                 ws += bfBSD;
  __bf16* vpb = (__bf16*)(ws);                 ws += bfBSD;
  __bf16* kpT = (__bf16*)(ws);                 ws += bfBSD;
  float* rmax = (float*)(ws);                  ws += (size_t)Bb * Hh * Ss * sizeof(float);
  float* rsum = (float*)(ws);

  // 1) convert inputs to bf16
  {
    int blocks = (int)(nBSD / 1024);                // 256 thr * 4 elem
    cvt_bf16_kernel<<<blocks, 256, 0, stream>>>(q, qb, (int)nBSD);
    cvt_bf16_kernel<<<blocks, 256, 0, stream>>>(k, kb, (int)nBSD);
    cvt_bf16_kernel<<<blocks, 256, 0, stream>>>(v, vb, (int)nBSD);
  }
  // 2) Wq transpose+convert
  wqt_kernel<<<(Dd * Dd) / 256, 256, 0, stream>>>(Wq, WqT);

  // 3) projections (shared Wq/bq, faithful to source quirk)
  {
    const int M = Bb * Ss;                          // 4096
    int waves = (M / 16) * (Dd / 16);               // 16384
    proj_wmma_kernel<<<waves / 4, 128, 0, stream>>>(qb, WqT, bq, qp, M);
    proj_wmma_kernel<<<waves / 4, 128, 0, stream>>>(kb, WqT, bq, kp, M);
    proj_wmma_kernel<<<waves / 4, 128, 0, stream>>>(vb, WqT, bq, vpb, M);
  }
  // 4) kp -> kpT
  kpt_kernel<<<(int)(nBSD / 256), 256, 0, stream>>>(kp, kpT);

  // 5) scores -> e (d_out)
  {
    int waves = Bb * Hh * (Ss / 16) * (Ss / 16);    // 524288
    scores_wmma_kernel<<<waves / 4, 128, 0, stream>>>(qp, kpT, kmask, e);
  }
  // 6) softmax stats
  softmax_stats_kernel<<<Bb * Hh * Ss, 256, 0, stream>>>(e, rmax, rsum);

  // 7) o = softmax(e) @ vp, gated by q_mask
  {
    int waves = Bb * Hh * (Ss / 16);                // 4096
    av_wmma_kernel<<<waves / 4, 128, 0, stream>>>(e, vpb, rmax, rsum, qmask, o);
  }
}
